// AttentionPooling_2826088481344
// MI455X (gfx1250) — compile-verified, matrix operand order checked by reading, not run on hardware
//
#include <hip/hip_runtime.h>
#include <hip/hip_bf16.h>
#include <cstdint>

// ---------------- problem constants ----------------
#define N_ROWS 262144
#define DIM    256
#define H_NUM  4
#define HD_NUM 64
#define B_SEG  8192

// ---------------- tiling ----------------
#define TILE    128              // rows per workgroup tile
#define NWAVES  8                // 256 threads, wave32
#define NKT     8                // K tiles of 32 (K = 256)
#define NCOLT   17               // 16 value col-tiles + 1 attn col-tile
#define BPACK_U32 (NCOLT * NKT * 256)   // 34816 dwords = 139264 B
#define NTILES  (N_ROWS / TILE)         // 2048
#define GRID_MAIN 512

typedef __bf16 bf16_t;
typedef bf16_t v16bf __attribute__((ext_vector_type(16)));
typedef float  v8f   __attribute__((ext_vector_type(8)));

union BfFrag { uint32_t u[8]; v16bf v; };

// f32 pair -> packed bf16 dword.  Round-to-nearest via +0x8000 on the raw
// bits, then merge the two high halves with one v_perm_b32.
__device__ __forceinline__ uint32_t bfpair(float a, float b) {
  uint32_t ua = __float_as_uint(a) + 0x8000u;
  uint32_t ub = __float_as_uint(b) + 0x8000u;
#if __has_builtin(__builtin_amdgcn_perm)
  // D.byte[0..3] = {ua.b2, ua.b3, ub.b2, ub.b3}  (sel codes: 0-3 -> s1, 4-7 -> s0)
  return __builtin_amdgcn_perm(ub, ua, 0x07060302u);
#else
  return (ua >> 16) | (ub & 0xFFFF0000u);
#endif
}

// ---------------- LDS layout for main kernel (bytes) ----------------
#define OFF_B     0
#define SZ_B      (BPACK_U32 * 4)     // 139264 : B fragments (bf16, swizzled)
#define OFF_V     (OFF_B + SZ_B)
#define SZ_V      (TILE * DIM * 4)    // 131072 : scaled V staging [128][256] f32
#define OFF_ATT   (OFF_V + SZ_V)
#define SZ_ATT    (TILE * H_NUM * 4)  //   2048 : attn_exp [128][4]
#define OFF_BATCH (OFF_ATT + SZ_ATT)
#define SZ_BATCH  ((TILE + 4) * 4)    //    528 : batch ids (+1 lookahead)
#define OFF_VB    (OFF_BATCH + SZ_BATCH)
#define SZ_VB     (DIM * 4)           //   1024 : value bias
#define OFF_BE    (OFF_VB + SZ_VB)
#define SZ_BE     16                  //        : attn bias (4 heads)
#define SMEM_MAIN (OFF_BE + SZ_BE)    // 273952 B  (< 320KB per WGP)
static_assert(SMEM_MAIN <= 320 * 1024, "LDS over budget");

// K-loop for one 16-col output tile with register double-buffered B
// fragments: ds_load of fragment kk+1 issues before WMMA kk, so LDS latency
// hides under the matrix op and no WAR v_nops are needed.
__device__ __forceinline__ v8f gemm_coltile(const char* bbase, const BfFrag* af) {
  v8f acc = {};
  v16bf cur = *(const v16bf*)(bbase);
  #pragma unroll
  for (int kk = 0; kk < NKT; ++kk) {
    v16bf nxt = (kk + 1 < NKT) ? *(const v16bf*)(bbase + (kk + 1) * 1024) : cur;
    acc = __builtin_amdgcn_wmma_f32_16x16x32_bf16(
        false, af[kk].v, false, cur, (short)0, acc, false, false);
    cur = nxt;
  }
  return acc;
}

// ====================================================================
// prep: q_eff = scale * key_w^T-projected query, b_eff, and the swizzled
// bf16 B-matrix fragment buffer: cols 0..255 = value_w^T, cols 256..259 =
// q_eff, cols 260..271 = 0.  Fragment layout mirrors the documented CDNA5
// 16-bit B striping: lane -> col (lane&15) / K-half (lane>>4), vgpr g ->
// K pair (g&3)*2 + (lane>>4)*8 + (g>>2)*16.
// ====================================================================
__global__ void prep_kernel(const float* __restrict__ query,
                            const float* __restrict__ key_w,
                            const float* __restrict__ key_b,
                            const float* __restrict__ value_w,
                            float* __restrict__ beff_g,
                            uint32_t* __restrict__ bpack_g) {
  __shared__ float qv[DIM];            // query [4*64]
  __shared__ float qe[H_NUM * DIM];    // q_eff [4][256]
  const int tid = threadIdx.x;
  const float scale = 0.125f;          // 64^-0.5

  qv[tid] = query[tid];
  __syncthreads();

  {
    const int j = tid;
    for (int h = 0; h < H_NUM; ++h) {
      float s = 0.f;
      for (int d = 0; d < HD_NUM; ++d)
        s += key_w[(size_t)(h * HD_NUM + d) * DIM + j] * qv[h * HD_NUM + d];
      qe[h * DIM + j] = s * scale;
    }
    if (tid < H_NUM) {
      float s = 0.f;
      for (int d = 0; d < HD_NUM; ++d)
        s += key_b[tid * HD_NUM + d] * qv[tid * HD_NUM + d];
      beff_g[tid] = s * scale;
    }
  }
  __syncthreads();

  for (int i = tid; i < BPACK_U32; i += blockDim.x) {
    const int frag = i >> 8;
    const int r    = i & 255;
    const int lane = r >> 3;
    const int g    = r & 7;
    const int c    = frag >> 3;    // col tile
    const int kk   = frag & 7;     // k tile
    const int col  = c * 16 + (lane & 15);
    const int kb   = kk * 32 + ((g & 3) * 2) + ((lane >> 4) * 8) + ((g >> 2) * 16);
    float e0, e1;
    if (col < DIM) {                       // B[k,c] = value_w[c,k]
      e0 = value_w[(size_t)col * DIM + kb];
      e1 = value_w[(size_t)col * DIM + kb + 1];
    } else if (col < DIM + H_NUM) {        // attn columns
      e0 = qe[(col - DIM) * DIM + kb];
      e1 = qe[(col - DIM) * DIM + kb + 1];
    } else { e0 = 0.f; e1 = 0.f; }
    bpack_g[i] = bfpair(e0, e1);
  }
}

// ====================================================================
// main: one streaming pass over x.  Per tile of 128 rows:
//  - pack per-wave A fragments (bf16) from global f32
//  - WMMA col-tile 16 -> attn logits -> clip/exp -> LDS
//  - WMMA col-tiles 0..15 -> (v + bias) * attn_exp -> LDS staging
//  - segmented reduce over sorted batch ids; flush via f32 atomics
// ====================================================================
__global__ void __launch_bounds__(256)
main_kernel(const float* __restrict__ x,
            const int* __restrict__ batch,
            const float* __restrict__ value_b,
            const float* __restrict__ beff_g,
            const uint32_t* __restrict__ bpack_g,
            float* __restrict__ raw_out,     // d_out (pre-zeroed): [B,256]
            float* __restrict__ seg_sum) {   // ws (pre-zeroed):   [B,4]
  extern __shared__ char smem[];
  uint32_t* ldsB   = (uint32_t*)(smem + OFF_B);
  float*    ldsV   = (float*)(smem + OFF_V);
  float*    ldsAtt = (float*)(smem + OFF_ATT);
  int*      ldsBat = (int*)(smem + OFF_BATCH);
  float*    ldsVb  = (float*)(smem + OFF_VB);
  float*    ldsBe  = (float*)(smem + OFF_BE);

  const int tid  = threadIdx.x;
  const int wave = tid >> 5;
  const int lane = tid & 31;
  const int l16  = lane & 15;
  const int lh   = lane >> 4;

  // stage B fragments + biases once per workgroup (L2-resident weights)
  {
    uint4* dst = (uint4*)ldsB;
    const uint4* src = (const uint4*)bpack_g;
    for (int i = tid; i < BPACK_U32 / 4; i += 256) dst[i] = src[i];
    ldsVb[tid] = value_b[tid];
    if (tid < H_NUM) ldsBe[tid] = beff_g[tid];
  }
  __syncthreads();

  for (int t = blockIdx.x; t < NTILES; t += gridDim.x) {
    const int base = t * TILE;
    if (tid <= TILE) {
      const int idx = base + tid;
      ldsBat[tid] = (idx < N_ROWS) ? batch[idx] : -1;
    }

    // ---- pack A fragments: 16 rows per wave, K = 256 ----
    const int row = base + wave * 16 + l16;
    const float* xr = x + (size_t)row * DIM + lh * 8;
    BfFrag af[NKT];
    #pragma unroll
    for (int kk = 0; kk < NKT; ++kk) {
      const float4* p = (const float4*)(xr + kk * 32);
      const float4 f0 = p[0], f1 = p[1], f2 = p[4], f3 = p[5];
      af[kk].u[0] = bfpair(f0.x, f0.y); af[kk].u[1] = bfpair(f0.z, f0.w);
      af[kk].u[2] = bfpair(f1.x, f1.y); af[kk].u[3] = bfpair(f1.z, f1.w);
      af[kk].u[4] = bfpair(f2.x, f2.y); af[kk].u[5] = bfpair(f2.z, f2.w);
      af[kk].u[6] = bfpair(f3.x, f3.y); af[kk].u[7] = bfpair(f3.z, f3.w);
    }

    // prefetch next grid-stride tile's x rows (global_prefetch_b8)
    if (t + (int)gridDim.x < NTILES) {
      const float* nxt = x + (size_t)(row + (int)gridDim.x * TILE) * DIM + lh * 8;
      __builtin_prefetch(nxt, 0, 0);
      __builtin_prefetch(nxt + 128, 0, 0);
    }

    const char* bwave = (const char*)ldsB + lane * 32;

    // ---- attn logits (col tile 16) ----
    {
      const v8f acc = gemm_coltile(bwave + 16 * NKT * 1024, af);
      if (l16 < H_NUM) {
        const float be = ldsBe[l16];
        #pragma unroll
        for (int g = 0; g < 8; ++g) {
          float aa = acc[g] + be;
          aa = fminf(fmaxf(aa, -20.f), 20.f);
          ldsAtt[(wave * 16 + lh * 8 + g) * H_NUM + l16] = __expf(aa);
        }
      }
    }

    // ---- value col tiles 0..15, scaled by attn_exp ----
    #pragma unroll 1
    for (int c = 0; c < 16; ++c) {
      const v8f acc = gemm_coltile(bwave + c * NKT * 1024, af);
      const float vb = ldsVb[c * 16 + l16];
      const int h = c >> 2;
      #pragma unroll
      for (int g = 0; g < 8; ++g) {
        const int rr = wave * 16 + lh * 8 + g;
        const float w = ldsAtt[rr * H_NUM + h];
        ldsV[rr * DIM + c * 16 + l16] = (acc[g] + vb) * w;
      }
    }
    __syncthreads();

    // ---- segmented reduce (batch is sorted; flush at boundaries) ----
    {
      float acc = 0.f, sacc = 0.f;
      const int d = tid;
      #pragma unroll 4
      for (int r = 0; r < TILE; ++r) {
        acc += ldsV[r * DIM + d];
        if (d < H_NUM) sacc += ldsAtt[r * H_NUM + d];
        const int b0 = ldsBat[r];
        if (r == TILE - 1 || b0 != ldsBat[r + 1]) {
          atomicAdd(&raw_out[(size_t)b0 * DIM + d], acc);
          acc = 0.f;
          if (d < H_NUM) { atomicAdd(&seg_sum[b0 * H_NUM + d], sacc); sacc = 0.f; }
        }
      }
    }
    __syncthreads();
  }
}

// out[b, h*64+d] = raw / (S[b,h] + 1e-8), in place on d_out
__global__ void finalize_kernel(float* __restrict__ out,
                                const float* __restrict__ seg_sum) {
  const int i = blockIdx.x * blockDim.x + threadIdx.x;
  const float s = seg_sum[(i >> 8) * H_NUM + ((i >> 6) & 3)];
  out[i] = out[i] / (s + 1e-8f);
}

__global__ void zero_kernel(float* __restrict__ p, int n) {
  const int i = blockIdx.x * blockDim.x + threadIdx.x;
  if (i < n) p[i] = 0.f;
}

// ====================================================================
extern "C" void kernel_launch(void* const* d_in, const int* in_sizes, int n_in,
                              void* d_out, int out_size, void* d_ws, size_t ws_size,
                              hipStream_t stream) {
  (void)in_sizes; (void)n_in; (void)out_size; (void)ws_size;
  const float* x       = (const float*)d_in[0];
  const int*   batch   = (const int*)d_in[1];
  const float* query   = (const float*)d_in[2];
  const float* key_w   = (const float*)d_in[3];
  const float* key_b   = (const float*)d_in[4];
  const float* value_w = (const float*)d_in[5];
  const float* value_b = (const float*)d_in[6];
  float* out = (float*)d_out;

  // workspace: [0,131072) seg_sum B*4 f32 | [131072,131088) b_eff | bpack 139264B
  float*    seg_sum = (float*)d_ws;
  float*    beff    = (float*)((char*)d_ws + 131072);
  uint32_t* bpack   = (uint32_t*)((char*)d_ws + 131088);

  zero_kernel<<<(B_SEG * DIM + 255) / 256, 256, 0, stream>>>(out, B_SEG * DIM);
  zero_kernel<<<(B_SEG * H_NUM + 255) / 256, 256, 0, stream>>>(seg_sum, B_SEG * H_NUM);
  prep_kernel<<<1, 256, 0, stream>>>(query, key_w, key_b, value_w, beff, bpack);

  (void)hipFuncSetAttribute((const void*)main_kernel,
                            hipFuncAttributeMaxDynamicSharedMemorySize, SMEM_MAIN);
  main_kernel<<<GRID_MAIN, 256, SMEM_MAIN, stream>>>(x, batch, value_b, beff,
                                                     bpack, out, seg_sum);
  finalize_kernel<<<B_SEG * DIM / 256, 256, 0, stream>>>(out, seg_sum);
}
